// custom_gnn_3831110828324
// MI455X (gfx1250) — matrix operand order repaired and checked by reading, not compiled, not run on hardware
//
#include <hip/hip_runtime.h>

#define DD 384            // feature dim
#define KP (DD/2)         // packed K pairs per weight matrix
#define NKT (DD/32)       // K-tiles per weight matrix (12)

// ---------- helpers ----------
__device__ __forceinline__ unsigned short f2bf(float f) {
    unsigned u = __float_as_uint(f);
    u += 0x7FFFu + ((u >> 16) & 1u);      // round-to-nearest-even
    return (unsigned short)(u >> 16);
}
__device__ __forceinline__ float bf2f(unsigned short h) {
    return __uint_as_float(((unsigned)h) << 16);
}
__device__ __forceinline__ unsigned pack2bf_hi(float f0, float f1,
                                               unsigned short& h0, unsigned short& h1) {
    h0 = f2bf(f0); h1 = f2bf(f1);
    return ((unsigned)h1 << 16) | h0;
}

typedef __attribute__((ext_vector_type(16))) __bf16 v16bf;
typedef __attribute__((ext_vector_type(8)))  float  v8f;

typedef union { unsigned int u[8]; uint4 q[2]; v16bf v; } abf_t;

// ---------- degree ----------
__global__ void sage_deg(const int* __restrict__ dst, float* __restrict__ deg, int nE) {
    int t = blockIdx.x * blockDim.x + threadIdx.x;
    if (t < nE) atomicAdd(&deg[dst[t]], 1.0f);
}
__global__ void sage_inv(float* __restrict__ deg, int n) {
    int t = blockIdx.x * blockDim.x + threadIdx.x;
    if (t < n) deg[t] = 1.0f / fmaxf(deg[t], 1.0f);
}

// ---------- pre-split weights into bf16 hi/lo, B-register order ----------
// WP[layer][kt][n][q] (half reads q = 4*half+j (j<4), 8+4*half+j-4 (j>=4), 16 u32 per (kt,n)):
//   pair k = kt*32 + 2q  ->  pack( bf16(W[k][n]), bf16(W[k+1][n]) )
__global__ void sage_wpack(const float* __restrict__ Wl, const float* __restrict__ Wr,
                           unsigned* __restrict__ Whl, unsigned* __restrict__ Wll,
                           unsigned* __restrict__ Whr, unsigned* __restrict__ Wlr,
                           int total) {
    int t = blockIdx.x * blockDim.x + threadIdx.x;
    if (t >= total) return;
    int layer = t / (KP * DD);
    int pi    = t - layer * (KP * DD);
    int kt    = pi / (DD * 16);
    int r     = pi - kt * (DD * 16);
    int n     = r >> 4;
    int q     = r & 15;
    int k     = kt * 32 + 2 * q;

    size_t sF = (size_t)layer * DD * DD + (size_t)k * DD + n;
    size_t sP = (size_t)layer * KP * DD + (size_t)kt * (DD * 16) + (size_t)n * 16 + q;

    unsigned short h0, h1;
    float a0 = Wl[sF], a1 = Wl[sF + DD];
    Whl[sP] = pack2bf_hi(a0, a1, h0, h1);
    Wll[sP] = pack2bf_hi(a0 - bf2f(h0), a1 - bf2f(h1), h0, h1);

    float b0 = Wr[sF], b1 = Wr[sF + DD];
    Whr[sP] = pack2bf_hi(b0, b1, h0, h1);
    Wlr[sP] = pack2bf_hi(b0 - bf2f(h0), b1 - bf2f(h1), h0, h1);
}

// ---------- edge scatter-add (mean numerator) ----------
__global__ void sage_scatter(const int* __restrict__ src, const int* __restrict__ dst,
                             const float* __restrict__ x, float* __restrict__ agg, int nE) {
    int t = blockIdx.x * blockDim.x + threadIdx.x;
    int total = nE * (DD / 4);
    if (t >= total) return;
    int e = t / (DD / 4);
    int c = (t - e * (DD / 4)) * 4;
    int s = src[e], d = dst[e];
    const float4 v = *(const float4*)(x + (size_t)s * DD + c);
    float* o = agg + (size_t)d * DD + c;
    atomicAdd(o + 0, v.x);
    atomicAdd(o + 1, v.y);
    atomicAdd(o + 2, v.z);
    atomicAdd(o + 3, v.w);
}

// ---------- fused SAGE layer GEMM:  out = (agg*invdeg)@W_l + x@W_r + b  [+relu] ----------
// bf16x3 split (hi*hi + hi*lo + lo*hi) on v_wmma_f32_16x16x32_bf16.
// Block = 8 waves, M-tile = 32 rows (2 strips) x all 384 cols.
// Each wave: 3 N-tiles x 2 strips = 6 accumulators, 18 WMMAs per K-step.
// A is split to hi/lo once cooperatively per K-step into LDS (ds_load_b128);
// the next step's global A fetch is software-pipelined under the WMMAs.
// Phase pointers are call-site constants so B stays on the GLOBAL load path.
template <bool RELU>
__global__ __launch_bounds__(256) void sage_gemm(
    const float* __restrict__ agg, const float* __restrict__ xin,
    const unsigned* __restrict__ Whl, const unsigned* __restrict__ Wll,
    const unsigned* __restrict__ Whr, const unsigned* __restrict__ Wlr,
    const float* __restrict__ bias, const float* __restrict__ invdeg,
    float* __restrict__ out, int nNodes) {
    // pair-packed bf16 A tiles; row stride 20 u32: 16B-aligned rows + conflict-free
    __shared__ unsigned AhiP[32][20];
    __shared__ unsigned AloP[32][20];

    const int tid  = threadIdx.x;
    const int lane = tid & 31;
    const int wave = tid >> 5;
    const int half = lane >> 4;    // lane group (0: lanes 0-15, 1: lanes 16-31)
    const int m    = lane & 15;    // A row / B,C column within tile
    const int rbase = blockIdx.x * 32;

    // staging role: thread -> rows (sr, sr+16), pair sp
    const int sr = tid >> 4;
    const int sp = tid & 15;

    int r0 = rbase + sr;      r0 = r0 < nNodes ? r0 : nNodes - 1;
    int r1 = rbase + 16 + sr; r1 = r1 < nNodes ? r1 : nNodes - 1;

    const float* ag0 = agg + (size_t)r0 * DD + 2 * sp;
    const float* ag1 = agg + (size_t)r1 * DD + 2 * sp;
    const float* ax0 = xin + (size_t)r0 * DD + 2 * sp;
    const float* ax1 = xin + (size_t)r1 * DD + 2 * sp;
    const float  id0 = invdeg[r0];
    const float  id1 = invdeg[r1];

    v8f acc[3][2];
    #pragma unroll
    for (int t = 0; t < 3; ++t)
        #pragma unroll
        for (int s = 0; s < 2; ++s)
            #pragma unroll
            for (int v = 0; v < 8; ++v) acc[t][s][v] = 0.0f;

    // pipeline registers (data + matching scale)
    float2 fa, fb;
    float  cs0, cs1;

    // phase body; all pointer args are literal at the two call sites
    auto run_phase = [&](const float* a0p, const float* a1p, float s0, float s1,
                         const unsigned* __restrict__ Bhi, const unsigned* __restrict__ Blo,
                         const float* na0p, const float* na1p, float ns0, float ns1,
                         bool hasNext) {
        #pragma unroll 1
        for (int kt = 0; kt < NKT; ++kt) {
            __syncthreads();
            {   // convert prefetched pairs -> LDS (only work between barriers)
                float a0 = fa.x * cs0, a1 = fa.y * cs0;
                float b0 = fb.x * cs1, b1 = fb.y * cs1;
                unsigned short h0, h1;
                AhiP[sr][sp]      = pack2bf_hi(a0, a1, h0, h1);
                AloP[sr][sp]      = pack2bf_hi(a0 - bf2f(h0), a1 - bf2f(h1), h0, h1);
                AhiP[sr + 16][sp] = pack2bf_hi(b0, b1, h0, h1);
                AloP[sr + 16][sp] = pack2bf_hi(b0 - bf2f(h0), b1 - bf2f(h1), h0, h1);
            }
            __syncthreads();

            // software-pipelined prefetch of the next step's A data
            if (kt + 1 < NKT) {
                fa = *(const float2*)(a0p + (kt + 1) * 32);
                fb = *(const float2*)(a1p + (kt + 1) * 32);
                cs0 = s0; cs1 = s1;
            } else if (hasNext) {
                fa = *(const float2*)na0p;
                fb = *(const float2*)na1p;
                cs0 = ns0; cs1 = ns1;
            }

            // A regs for both strips: two contiguous b128 chunks per operand
            abf_t ah[2], al[2];
            #pragma unroll
            for (int s = 0; s < 2; ++s) {
                ah[s].q[0] = *(const uint4*)&AhiP[s * 16 + m][4 * half];
                ah[s].q[1] = *(const uint4*)&AhiP[s * 16 + m][8 + 4 * half];
                al[s].q[0] = *(const uint4*)&AloP[s * 16 + m][4 * half];
                al[s].q[1] = *(const uint4*)&AloP[s * 16 + m][8 + 4 * half];
            }

            // B regs for the wave's 3 N-tiles
            abf_t bh[3], bl[3];
            #pragma unroll
            for (int t = 0; t < 3; ++t) {
                int n = (wave * 3 + t) * 16 + m;
                size_t off = ((size_t)kt * DD + n) * 16 + 4 * half;
                bh[t].q[0] = *(const uint4*)(Bhi + off);
                bh[t].q[1] = *(const uint4*)(Bhi + off + 8);
                bl[t].q[0] = *(const uint4*)(Blo + off);
                bl[t].q[1] = *(const uint4*)(Blo + off + 8);
            }

            // 18 WMMAs, 6 independent accumulator chains interleaved
            #pragma unroll
            for (int t = 0; t < 3; ++t)
                #pragma unroll
                for (int s = 0; s < 2; ++s)
                    acc[t][s] = __builtin_amdgcn_wmma_f32_16x16x32_bf16(
                        false, ah[s].v, false, bh[t].v, (short)0, acc[t][s], false, false);
            #pragma unroll
            for (int t = 0; t < 3; ++t)
                #pragma unroll
                for (int s = 0; s < 2; ++s)
                    acc[t][s] = __builtin_amdgcn_wmma_f32_16x16x32_bf16(
                        false, ah[s].v, false, bl[t].v, (short)0, acc[t][s], false, false);
            #pragma unroll
            for (int t = 0; t < 3; ++t)
                #pragma unroll
                for (int s = 0; s < 2; ++s)
                    acc[t][s] = __builtin_amdgcn_wmma_f32_16x16x32_bf16(
                        false, al[s].v, false, bh[t].v, (short)0, acc[t][s], false, false);
        }
    };

    // prime the pipeline with phase-0 step 0
    fa = *(const float2*)ag0;
    fb = *(const float2*)ag1;
    cs0 = id0; cs1 = id1;

    run_phase(ag0, ag1, id0, id1, Whl, Wll, ax0, ax1, 1.0f, 1.0f, true);
    run_phase(ax0, ax1, 1.0f, 1.0f, Whr, Wlr, nullptr, nullptr, 0.0f, 0.0f, false);

    // Epilogue: bias (+relu). C layout: VGPR v -> row = v + 8*half, col = lane&15.
    // Strip guard is uniform across the block; fast path has no exec juggling.
    #pragma unroll
    for (int s = 0; s < 2; ++s) {
        const int sb = rbase + s * 16;
        const bool fullStrip = (sb + 16) <= nNodes;
        #pragma unroll
        for (int t = 0; t < 3; ++t) {
            int n = (wave * 3 + t) * 16 + m;
            float b = bias[n];
            if (fullStrip) {
                #pragma unroll
                for (int v = 0; v < 8; ++v) {
                    float val = acc[t][s][v] + b;
                    if (RELU) val = fmaxf(val, 0.0f);
                    out[(size_t)(sb + v + 8 * half) * DD + n] = val;
                }
            } else {
                #pragma unroll
                for (int v = 0; v < 8; ++v) {
                    int row = sb + v + 8 * half;
                    if (row < nNodes) {
                        float val = acc[t][s][v] + b;
                        if (RELU) val = fmaxf(val, 0.0f);
                        out[(size_t)row * DD + n] = val;
                    }
                }
            }
        }
    }
}

// ---------- launcher ----------
extern "C" void kernel_launch(void* const* d_in, const int* in_sizes, int n_in,
                              void* d_out, int out_size, void* d_ws, size_t ws_size,
                              hipStream_t stream) {
    (void)n_in; (void)out_size; (void)ws_size;
    const float* emb = (const float*)d_in[0];
    const int*   ei  = (const int*)d_in[1];
    const float* Wl  = (const float*)d_in[2];
    const float* Wr  = (const float*)d_in[3];
    const float* bl  = (const float*)d_in[4];
    float* out = (float*)d_out;
    char*  ws  = (char*)d_ws;

    const int N = in_sizes[0] / DD;          // 50000
    const int E = in_sizes[1] / 2;           // 400000
    const int L = in_sizes[2] / (DD * DD);   // 4

    // workspace layout
    const size_t O_INV = 0;                              // N floats
    const size_t O_WHL = 262144;                         // packed W blocks: L*KP*DD u32 each
    const size_t WBLK  = (size_t)L * KP * DD * 4;
    const size_t O_WLL = O_WHL + WBLK;
    const size_t O_WHR = O_WLL + WBLK;
    const size_t O_WLR = O_WHR + WBLK;
    const size_t O_AGG = ((O_WLR + WBLK + 0xFFFFF) / 0x100000) * 0x100000;
    const size_t XB    = (size_t)N * DD * sizeof(float);
    const size_t O_PING = O_AGG + XB;

    float*    invdeg = (float*)(ws + O_INV);
    unsigned* Whl = (unsigned*)(ws + O_WHL);
    unsigned* Wll = (unsigned*)(ws + O_WLL);
    unsigned* Whr = (unsigned*)(ws + O_WHR);
    unsigned* Wlr = (unsigned*)(ws + O_WLR);
    float*    aggb = (float*)(ws + O_AGG);
    float*    ping = (float*)(ws + O_PING);

    const int* srcIdx = ei;
    const int* dstIdx = ei + E;

    // degree -> inv_deg
    hipMemsetAsync(invdeg, 0, (size_t)N * sizeof(float), stream);
    sage_deg<<<(E + 255) / 256, 256, 0, stream>>>(dstIdx, invdeg, E);
    sage_inv<<<(N + 255) / 256, 256, 0, stream>>>(invdeg, N);

    // split weights into bf16 hi/lo, packed in B-register order
    const int wtotal = L * KP * DD;
    sage_wpack<<<(wtotal + 255) / 256, 256, 0, stream>>>(Wl, Wr, Whl, Wll, Whr, Wlr, wtotal);

    // ping-pong so the final layer lands in d_out
    float* bufs[2];
    if (L % 2 == 0) { bufs[0] = ping; bufs[1] = out; }
    else            { bufs[0] = out;  bufs[1] = ping; }

    const float* xcur = emb;
    const int perlayer = KP * DD;  // u32 stride per layer in packed W
    const int gemmGrid = (N + 31) / 32;
    for (int layer = 0; layer < L; ++layer) {
        hipMemsetAsync(aggb, 0, XB, stream);
        int sthreads = E * (DD / 4);
        sage_scatter<<<(sthreads + 255) / 256, 256, 0, stream>>>(srcIdx, dstIdx, xcur, aggb, E);

        float* xnext = bufs[layer & 1];
        if (layer < L - 1) {
            sage_gemm<true><<<gemmGrid, 256, 0, stream>>>(
                aggb, xcur,
                Whl + (size_t)layer * perlayer, Wll + (size_t)layer * perlayer,
                Whr + (size_t)layer * perlayer, Wlr + (size_t)layer * perlayer,
                bl + (size_t)layer * DD, invdeg, xnext, N);
        } else {
            sage_gemm<false><<<gemmGrid, 256, 0, stream>>>(
                aggb, xcur,
                Whl + (size_t)layer * perlayer, Wll + (size_t)layer * perlayer,
                Whr + (size_t)layer * perlayer, Wlr + (size_t)layer * perlayer,
                bl + (size_t)layer * DD, invdeg, xnext, N);
        }
        xcur = xnext;
    }
}
